// SparseAttention_14680198218474
// MI455X (gfx1250) — matrix operand order repaired and checked by reading, not compiled, not run on hardware
//
#include <hip/hip_runtime.h>

// ---------------------------------------------------------------------------
// Flash-attention for B=2, H=12, S=2048, D=64 (fp32 in/out) on gfx1250.
// QK^T and PV via v_wmma_f32_16x16x32_f16 (fp32 accumulate), 64-key chunks.
// Workgroup: 128 threads (4 wave32s), 64 q-rows per WG, 16 q-rows per wave.
// ---------------------------------------------------------------------------

typedef __attribute__((ext_vector_type(16))) _Float16 v16h;
typedef __attribute__((ext_vector_type(8)))  _Float16 v8h;
typedef __attribute__((ext_vector_type(4)))  _Float16 v4h;
typedef __attribute__((ext_vector_type(8)))  float    v8f;
typedef __attribute__((ext_vector_type(4)))  float    v4f;

#define S_LEN 2048
#define NHEAD 12
#define NBATCH 2
#define HDIM 64
#define QTILE 64   // q rows per workgroup
#define KTILE 64   // keys per inner chunk (2 WMMA K-steps)
#define NTHREADS 128

#define WMMA_F16(a, b, c) \
  __builtin_amdgcn_wmma_f32_16x16x32_f16(false, (a), false, (b), (short)0, (c), false, false)

// A-type fragment (16 x 32, f16). Contiguous layout:
// lane m = lane%16; VGPR0-7 hold K = k0..k0+15,
// where k0 = 0 (lanes 0-15) or 16 (lanes 16-31).
__device__ __forceinline__ v16h load_a_frag(const _Float16* base, int rowStride) {
  const int lane = threadIdx.x & 31;
  const int m    = lane & 15;
  const int k0   = (lane < 16) ? 0 : 16;
  const _Float16* p = base + m * rowStride + k0;
  v8h lo = *(const v8h*)(p);        // K = k0 .. k0+7
  v8h hi = *(const v8h*)(p + 8);    // K = k0+8 .. k0+15
  v16h r;
#pragma unroll
  for (int i = 0; i < 8; ++i) { r[i] = lo[i]; r[i + 8] = hi[i]; }
  return r;
}

__global__ __launch_bounds__(NTHREADS)
void fattn_wmma_kernel(const float* __restrict__ Qg,
                       const float* __restrict__ Kg,
                       const float* __restrict__ Vg,
                       const int*   __restrict__ Mask,
                       float*       __restrict__ Out) {
  // LDS (f16): Q tile, K chunk (row-major [key][d]),
  // V chunk transposed+permuted ([d][pos]), P scratch per wave ([row][pos]).
  // Permutation of the PV reduction dim: position j <-> key (16*(j&3) + (j>>2)).
  // P columns and V rows use the same sigma, so P·V is unchanged, but each
  // lane stores its 4 probabilities as one contiguous ds_store_b64.
  __shared__ _Float16 Qs[QTILE * HDIM];        // 8 KB
  __shared__ _Float16 Ks[KTILE * HDIM];        // 8 KB
  __shared__ _Float16 Vt[HDIM * KTILE];        // 8 KB
  __shared__ _Float16 Ps[4 * 16 * KTILE];      // 8 KB

  const int tid  = threadIdx.x;
  const int lane = tid & 31;
  const int wave = tid >> 5;
  const int ln   = lane & 15;     // N (or M) index within 16
  const int hi16 = lane >> 4;     // lane-half select

  const int nQ = S_LEN / QTILE;   // 32
  const int bh = blockIdx.x / nQ;
  const int qt = blockIdx.x % nQ;

  const size_t bh_off = (size_t)bh * S_LEN * HDIM;
  const float* Qp = Qg + bh_off + (size_t)qt * QTILE * HDIM;
  const float* Kp = Kg + bh_off;
  const float* Vp = Vg + bh_off;

  // ---- stage Q tile as f16 (b128 loads, b64 LDS stores), scale by 0.125 ---
  for (int i4 = tid * 4; i4 < QTILE * HDIM; i4 += NTHREADS * 4) {
    const v4f q = *(const v4f*)(Qp + i4);
    v4h h;
#pragma unroll
    for (int j = 0; j < 4; ++j) h[j] = (_Float16)(q[j] * 0.125f);
    *(v4h*)&Qs[i4] = h;
  }
  __syncthreads();

  const _Float16* qbase = &Qs[wave * 16 * HDIM];
  const v16h aq0 = load_a_frag(qbase,      HDIM);   // d = 0..31
  const v16h aq1 = load_a_frag(qbase + 32, HDIM);   // d = 32..63

  // C/D layout: VGPR r <-> row (r + 8*hi16), lane%16 <-> col
  v8f o0 = {}, o1 = {}, o2 = {}, o3 = {};
  float mrow[8], lrow[8];
#pragma unroll
  for (int r = 0; r < 8; ++r) { mrow[r] = -__builtin_inff(); lrow[r] = 0.0f; }

  const int qrow0 = qt * QTILE + wave * 16 + 8 * hi16;  // row for VGPR r is qrow0+r
  const int* mp[8];
#pragma unroll
  for (int r = 0; r < 8; ++r) mp[r] = Mask + (size_t)(qrow0 + r) * S_LEN;

  // Per-lane fragment pointers (loop-invariant).
  const _Float16* kp0 = &Ks[(size_t)(ln)      * HDIM + 16 * hi16];
  const _Float16* kp1 = &Ks[(size_t)(16 + ln) * HDIM + 16 * hi16];
  const _Float16* kp2 = &Ks[(size_t)(32 + ln) * HDIM + 16 * hi16];
  const _Float16* kp3 = &Ks[(size_t)(48 + ln) * HDIM + 16 * hi16];
  const _Float16* vp  = &Vt[(size_t)ln * KTILE + 16 * hi16];
  const _Float16* pbase = &Ps[(size_t)wave * 16 * KTILE];

  for (int kb = 0; kb < S_LEN; kb += KTILE) {
    __syncthreads();
    // ---- stage K chunk (row-major) and V chunk (transposed+permuted) -----
    {
      const float* kc = Kp + (size_t)kb * HDIM;
      const float* vc = Vp + (size_t)kb * HDIM;
      for (int i4 = tid * 4; i4 < KTILE * HDIM; i4 += NTHREADS * 4) {
        const v4f kv = *(const v4f*)(kc + i4);
        v4h kh;
#pragma unroll
        for (int j = 0; j < 4; ++j) kh[j] = (_Float16)kv[j];
        *(v4h*)&Ks[i4] = kh;

        const v4f vv = *(const v4f*)(vc + i4);
        const int r  = i4 >> 6;                       // key within chunk
        const int c0 = i4 & 63;                       // starting dim
        const int pj = ((r & 15) << 2) | (r >> 4);    // permuted position
#pragma unroll
        for (int j = 0; j < 4; ++j)
          Vt[(c0 + j) * KTILE + pj] = (_Float16)vv[j];
      }
    }
    __syncthreads();

    // ---- prefetch next chunk into cache while this chunk computes --------
    if (kb + KTILE < S_LEN) {
      const char* nk = (const char*)(Kp + (size_t)(kb + KTILE) * HDIM);
      const char* nv = (const char*)(Vp + (size_t)(kb + KTILE) * HDIM);
      __builtin_prefetch(nk + tid * 128, 0, 1);
      __builtin_prefetch(nv + tid * 128, 0, 1);
    }

    // ---- scores S = Q K^T : 16x64 (four 16x16 tiles), K-dim = 64 ---------
    // Distinct registers per fragment so DS loads pipeline ahead of WMMAs.
    v8f s0 = {}, s1 = {}, s2 = {}, s3 = {};
    {
      v16h b0 = *(const v16h*)(kp0);
      v16h b1 = *(const v16h*)(kp1);
      v16h b2 = *(const v16h*)(kp2);
      v16h b3 = *(const v16h*)(kp3);
      s0 = WMMA_F16(aq0, b0, s0);
      s1 = WMMA_F16(aq0, b1, s1);
      s2 = WMMA_F16(aq0, b2, s2);
      s3 = WMMA_F16(aq0, b3, s3);
      b0 = *(const v16h*)(kp0 + 32);
      b1 = *(const v16h*)(kp1 + 32);
      b2 = *(const v16h*)(kp2 + 32);
      b3 = *(const v16h*)(kp3 + 32);
      s0 = WMMA_F16(aq1, b0, s0);
      s1 = WMMA_F16(aq1, b1, s1);
      s2 = WMMA_F16(aq1, b2, s2);
      s3 = WMMA_F16(aq1, b3, s3);
    }

    // ---- mask (mask==0 -> -inf); all-ones here, L2-resident --------------
#pragma unroll
    for (int r = 0; r < 8; ++r) {
      if (mp[r][kb + ln] == 0)      s0[r] = -__builtin_inff();
      if (mp[r][kb + 16 + ln] == 0) s1[r] = -__builtin_inff();
      if (mp[r][kb + 32 + ln] == 0) s2[r] = -__builtin_inff();
      if (mp[r][kb + 48 + ln] == 0) s3[r] = -__builtin_inff();
    }

    // ---- online softmax (row stats via xor-shuffles within 16-lane half) -
#pragma unroll
    for (int r = 0; r < 8; ++r) {
      float x = fmaxf(fmaxf(s0[r], s1[r]), fmaxf(s2[r], s3[r]));
      x = fmaxf(x, __shfl_xor(x, 1, 32));
      x = fmaxf(x, __shfl_xor(x, 2, 32));
      x = fmaxf(x, __shfl_xor(x, 4, 32));
      x = fmaxf(x, __shfl_xor(x, 8, 32));
      const float mn = fmaxf(mrow[r], x);
      const float p0 = __expf(s0[r] - mn);
      const float p1 = __expf(s1[r] - mn);
      const float p2 = __expf(s2[r] - mn);
      const float p3 = __expf(s3[r] - mn);
      float rs = (p0 + p1) + (p2 + p3);
      rs += __shfl_xor(rs, 1, 32);
      rs += __shfl_xor(rs, 2, 32);
      rs += __shfl_xor(rs, 4, 32);
      rs += __shfl_xor(rs, 8, 32);
      const float sc = __expf(mrow[r] - mn);
      lrow[r] = lrow[r] * sc + rs;
      mrow[r] = mn;
      o0[r] *= sc; o1[r] *= sc; o2[r] *= sc; o3[r] *= sc;
      // one contiguous 8B store of {p0,p1,p2,p3} at permuted position ln*4
      v4h pk4;
      pk4[0] = (_Float16)p0; pk4[1] = (_Float16)p1;
      pk4[2] = (_Float16)p2; pk4[3] = (_Float16)p3;
      *(v4h*)&Ps[(size_t)(wave * 16 + r + 8 * hi16) * KTILE + ln * 4] = pk4;
    }

    // ---- O += P V : A = P(16x64) from LDS (two K-steps), B via Vt --------
    {
      const v16h ap0 = load_a_frag(pbase,      KTILE);   // positions 0..31
      const v16h ap1 = load_a_frag(pbase + 32, KTILE);   // positions 32..63
      v16h b0 = *(const v16h*)(vp);
      v16h b1 = *(const v16h*)(vp + 16 * KTILE);
      v16h b2 = *(const v16h*)(vp + 32 * KTILE);
      v16h b3 = *(const v16h*)(vp + 48 * KTILE);
      o0 = WMMA_F16(ap0, b0, o0);
      o1 = WMMA_F16(ap0, b1, o1);
      o2 = WMMA_F16(ap0, b2, o2);
      o3 = WMMA_F16(ap0, b3, o3);
      b0 = *(const v16h*)(vp + 32);
      b1 = *(const v16h*)(vp + 16 * KTILE + 32);
      b2 = *(const v16h*)(vp + 32 * KTILE + 32);
      b3 = *(const v16h*)(vp + 48 * KTILE + 32);
      o0 = WMMA_F16(ap1, b0, o0);
      o1 = WMMA_F16(ap1, b1, o1);
      o2 = WMMA_F16(ap1, b2, o2);
      o3 = WMMA_F16(ap1, b3, o3);
    }
  }

  // ---- epilogue: O / l, write fp32 ---------------------------------------
  float* og = Out + bh_off + (size_t)qt * QTILE * HDIM + (size_t)wave * 16 * HDIM;
#pragma unroll
  for (int r = 0; r < 8; ++r) {
    const float inv = 1.0f / lrow[r];
    float* orow = og + (size_t)(r + 8 * hi16) * HDIM + ln;
    orow[0]  = o0[r] * inv;
    orow[16] = o1[r] * inv;
    orow[32] = o2[r] * inv;
    orow[48] = o3[r] * inv;
  }
}

extern "C" void kernel_launch(void* const* d_in, const int* in_sizes, int n_in,
                              void* d_out, int out_size, void* d_ws, size_t ws_size,
                              hipStream_t stream) {
  (void)in_sizes; (void)n_in; (void)out_size; (void)d_ws; (void)ws_size;
  const float* Q = (const float*)d_in[0];
  const float* K = (const float*)d_in[1];
  const float* V = (const float*)d_in[2];
  const int*   M = (const int*)d_in[3];
  float* O = (float*)d_out;

  const dim3 grid(NBATCH * NHEAD * (S_LEN / QTILE));  // 768 workgroups
  fattn_wmma_kernel<<<grid, NTHREADS, 0, stream>>>(Q, K, V, M, O);
}
